// ScaledDotProductAttention_47175920779814
// MI455X (gfx1250) — matrix operand order repaired and checked by reading, not compiled, not run on hardware
//
#include <hip/hip_runtime.h>
#include <hip/hip_bf16.h>
#include <math.h>

typedef __attribute__((ext_vector_type(16))) _Float16 v16h;
typedef __attribute__((ext_vector_type(8)))  _Float16 v8h;
typedef __attribute__((ext_vector_type(8)))  float    v8f;
typedef __attribute__((ext_vector_type(4)))  float    v4f;
typedef __attribute__((ext_vector_type(4)))  int      v4i;

#define BB   16
#define LQ   2048
#define LK   2048
#define DD   64
#define QT   16            // queries per workgroup
#define NEGINF (-1e9f)

// padded LDS row strides (bank-conflict-free b128 fragment loads)
#define SS_STRIDE 2052     // floats  (+4): bank start = 4*m mod 64, distinct
#define SQ_STRIDE 72       // f16     (+8): bank start = 36*m mod 64, distinct

// dynamic LDS layout (bytes) -- total 137792 B => 2 workgroups per WGP
#define OFF_SS   0                               // float [16][2052] = 131328
#define OFF_SQ   (OFF_SS + 16*SS_STRIDE*4)       // f16   [16][72]   =   2304
#define OFF_PART (OFF_SQ + 16*SQ_STRIDE*2)       // float [4][16][16]=   4096
#define OFF_INV  (OFF_PART + 4*16*16*4)          // float [16]       =     64
#define LDS_BYTES (OFF_INV + 16*4)

// wave32 XOR butterfly via ds_swizzle_b32 (group-of-32: {xor<<10 | or<<5 | and})
#define SWZ_F(x, imm) __int_as_float(__builtin_amdgcn_ds_swizzle(__float_as_int(x), (imm)))
#define WAVE_RED(var, OPEXPR)                                   \
    do {                                                        \
        float _o;                                               \
        _o = SWZ_F(var, 0x401f); var = OPEXPR; /* xor 16 */     \
        _o = SWZ_F(var, 0x201f); var = OPEXPR; /* xor 8  */     \
        _o = SWZ_F(var, 0x101f); var = OPEXPR; /* xor 4  */     \
        _o = SWZ_F(var, 0x081f); var = OPEXPR; /* xor 2  */     \
        _o = SWZ_F(var, 0x041f); var = OPEXPR; /* xor 1  */     \
    } while (0)

__global__ __launch_bounds__(256)
void sdpa_fused_kernel(const float* __restrict__ q,
                       const float* __restrict__ k,
                       const float* __restrict__ v,
                       const int*   __restrict__ mask,
                       float* __restrict__ out,
                       float* __restrict__ attn)
{
    extern __shared__ char smem[];
    float*    sS   = (float*)(smem + OFF_SS);
    _Float16* sQ   = (_Float16*)(smem + OFF_SQ);
    float*    sPt  = (float*)(smem + OFF_PART);
    float*    sInv = (float*)(smem + OFF_INV);

    const int blk  = blockIdx.x;            // 0 .. B*(LQ/16)-1
    const int b    = blk >> 7;              // LQ/16 = 128 q-tiles per batch
    const int q0   = (blk & 127) * QT;

    const int tid  = threadIdx.x;
    const int wave = tid >> 5;
    const int lane = tid & 31;
    const int m    = lane & 15;
    const int hi   = (lane >> 4) & 1;

    const float* qb = q + ((size_t)b * LQ + q0) * DD;
    const float* kb = k + (size_t)b * LK * DD;
    const float* vb = v + (size_t)b * LK * DD;

    // ---------------- stage Q tile to LDS as f16 (padded rows) ---------------
    for (int i = tid; i < QT * DD; i += 256)
        sQ[(i >> 6) * SQ_STRIDE + (i & 63)] = (_Float16)qb[(i >> 6) * DD + (i & 63)];
    __syncthreads();

    // ---- A fragments for Q: per lane two contiguous 8-half runs per chunk ---
    // k-map: e=0..7 -> k = hi*8 + e ; e=8..15 -> k = 16 + hi*8 + (e-8)
    v16h aQ0, aQ1;
    {
        const _Float16* qrow = sQ + m * SQ_STRIDE + hi * 8;
        v8h l0 = *(const v8h*)(qrow +  0);
        v8h h0 = *(const v8h*)(qrow + 16);
        v8h l1 = *(const v8h*)(qrow + 32);
        v8h h1 = *(const v8h*)(qrow + 48);
#pragma unroll
        for (int e = 0; e < 8; ++e) {
            aQ0[e] = l0[e];  aQ0[8 + e] = h0[e];
            aQ1[e] = l1[e];  aQ1[8 + e] = h1[e];
        }
    }

    // ---------------- Phase 1: S = Q K^T / sqrt(D) -> LDS --------------------
    // each wave owns 256 k-columns = 16 tiles of 16x16
    for (int t = 0; t < 16; ++t) {
        const int n0 = wave * 256 + t * 16;
        const float* krow = kb + (size_t)(n0 + m) * DD + hi * 8;
        v4f kv[8];
#pragma unroll
        for (int t4 = 0; t4 < 4; ++t4) kv[t4]     = *(const v4f*)(krow + 4 * (t4 & 1) + 16 * (t4 >> 1));
#pragma unroll
        for (int t4 = 0; t4 < 4; ++t4) kv[4 + t4] = *(const v4f*)(krow + 32 + 4 * (t4 & 1) + 16 * (t4 >> 1));
        v16h b0, b1;
#pragma unroll
        for (int e = 0; e < 16; ++e) {
            b0[e] = (_Float16)kv[e >> 2][e & 3];
            b1[e] = (_Float16)kv[4 + (e >> 2)][e & 3];
        }
        v8f c = {};
        c = __builtin_amdgcn_wmma_f32_16x16x32_f16(false, aQ0, false, b0,
                                                   (short)0, c, false, false);
        c = __builtin_amdgcn_wmma_f32_16x16x32_f16(false, aQ1, false, b1,
                                                   (short)0, c, false, false);
#pragma unroll
        for (int i = 0; i < 8; ++i)
            sS[(i + hi * 8) * SS_STRIDE + n0 + m] = c[i] * 0.125f;  // 1/sqrt(64)
    }
    __syncthreads();

    // ---------------- Phase 2: masked softmax (fp32), write attn -------------
    for (int rr = 0; rr < 2; ++rr) {
        const int r = wave * 2 + rr;
        float* srow = sS + r * SS_STRIDE;
        const int* mrow = mask + ((size_t)b * LQ + q0 + r) * (size_t)LK;

        // pass A: apply additive mask, row max (vectorized x4)
        float mx = -INFINITY;
        for (int it = 0; it < LK / 128; ++it) {
            const int j = it * 128 + lane * 4;
            v4i mm = *(const v4i*)(mrow + j);
            v4f ss = *(v4f*)(srow + j);
#pragma unroll
            for (int cc = 0; cc < 4; ++cc) {
                ss[cc] += (mm[cc] ? 0.0f : NEGINF);
                mx = fmaxf(mx, ss[cc]);
            }
            *(v4f*)(srow + j) = ss;
        }
        WAVE_RED(mx, fmaxf(mx, _o));

        // pass B: p = exp(s - mx); overwrite srow with p (fp32)
        float sum = 0.0f;
        for (int it = 0; it < LK / 128; ++it) {
            const int j = it * 128 + lane * 4;
            v4f ss = *(v4f*)(srow + j);
            v4f pp;
#pragma unroll
            for (int cc = 0; cc < 4; ++cc) {
                pp[cc] = __expf(ss[cc] - mx);
                sum += pp[cc];
            }
            *(v4f*)(srow + j) = pp;
        }
        WAVE_RED(sum, sum + _o);

        const float inv = 1.0f / sum;
        if (lane == 0) sInv[r] = inv;

        // pass C: attn = p * inv (fp32-exact, vectorized store)
        float* arow = attn + ((size_t)b * LQ + q0 + r) * (size_t)LK;
        for (int it = 0; it < LK / 128; ++it) {
            const int j = it * 128 + lane * 4;
            v4f pp = *(const v4f*)(srow + j);
#pragma unroll
            for (int cc = 0; cc < 4; ++cc) pp[cc] *= inv;
            *(v4f*)(arow + j) = pp;
        }
    }
    __syncthreads();

    // ---------------- Phase 3: O = (P V) * inv_sum ---------------------------
    // waves 0..3 -> d-tile 0..3, k-half 0 ; waves 4..7 -> d-tile 0..3, k-half 1
    const int ntile = wave & 3;
    const int n0    = ntile * 16;
    const int khalf = wave >> 2;

    v8f o = {};
    for (int kc = 0; kc < 32; ++kc) {
        const int k0 = khalf * 1024 + kc * 32;

        // A fragment from fp32 p in sS: 4 conflict-free b128 LDS loads + cvt
        const float* prow = sS + m * SS_STRIDE + k0 + hi * 8;
        v4f pa = *(const v4f*)(prow);        // k = hi*8 + 0..3
        v4f pb = *(const v4f*)(prow + 4);    // k = hi*8 + 4..7
        v4f pc = *(const v4f*)(prow + 16);   // k = 16 + hi*8 + 0..3
        v4f pd = *(const v4f*)(prow + 20);   // k = 16 + hi*8 + 4..7
        v16h a;
#pragma unroll
        for (int e = 0; e < 4; ++e) {
            a[e]      = (_Float16)pa[e];
            a[4 + e]  = (_Float16)pb[e];
            a[8 + e]  = (_Float16)pc[e];
            a[12 + e] = (_Float16)pd[e];
        }

        // B fragment from global V: lane = column n0+m, k = k0 + hi*8 + e (e<8), k0 + 16 + hi*8 + (e-8) (e>=8)
        const float* vcol = vb + (size_t)(k0 + hi * 8) * DD + n0 + m;
        v16h bf;
#pragma unroll
        for (int e = 0; e < 16; ++e)
            bf[e] = (_Float16)vcol[(size_t)(e + (e >> 3) * 8) * DD];

        o = __builtin_amdgcn_wmma_f32_16x16x32_f16(false, a, false, bf,
                                                   (short)0, o, false, false);
    }

    // cross-k-half reduction via LDS, then scale + store fp32 output
    if (khalf == 1) {
#pragma unroll
        for (int i = 0; i < 8; ++i)
            sPt[ntile * 256 + (i + hi * 8) * 16 + m] = o[i];
    }
    __syncthreads();
    if (khalf == 0) {
        float* ob = out + ((size_t)b * LQ + q0) * DD;
#pragma unroll
        for (int i = 0; i < 8; ++i) {
            const int row = i + hi * 8;
            const float val = (o[i] + sPt[ntile * 256 + row * 16 + m]) * sInv[row];
            ob[(size_t)row * DD + n0 + m] = val;
        }
    }
}

extern "C" void kernel_launch(void* const* d_in, const int* in_sizes, int n_in,
                              void* d_out, int out_size, void* d_ws, size_t ws_size,
                              hipStream_t stream) {
    (void)in_sizes; (void)n_in; (void)out_size; (void)d_ws; (void)ws_size;
    const float* q    = (const float*)d_in[0];
    const float* k    = (const float*)d_in[1];
    const float* v    = (const float*)d_in[2];
    const int*   mask = (const int*)d_in[3];

    float* out  = (float*)d_out;                         // [16,2048,64]
    float* attn = out + (size_t)BB * LQ * DD;            // [16,2048,2048]

    const int grid = BB * (LQ / QT);                     // 2048 workgroups
    sdpa_fused_kernel<<<grid, 256, LDS_BYTES, stream>>>(q, k, v, mask, out, attn);
}
